// AttnDecoder_7267084665224
// MI455X (gfx1250) — matrix operand order repaired and checked by reading, not compiled, not run on hardware
//
#include <hip/hip_runtime.h>
#include <math.h>

// Problem constants (match reference)
#define BB   128
#define SS   128
#define EE   256
#define DD   256
#define OUTN 64
#define H4   1024  // 4*D

typedef __attribute__((ext_vector_type(16))) __bf16 v16bf;
typedef __attribute__((ext_vector_type(8)))  float  v8f;

// ---------- scalar helpers ----------
__device__ __forceinline__ unsigned short f2bf(float f) {
  unsigned u = __builtin_bit_cast(unsigned, f);
  u += 0x7FFFu + ((u >> 16) & 1u);
  return (unsigned short)(u >> 16);
}
__device__ __forceinline__ float bf2f(unsigned short h) {
  unsigned u = ((unsigned)h) << 16;
  return __builtin_bit_cast(float, u);
}
__device__ __forceinline__ __bf16 bfbits(unsigned short h) {
  return __builtin_bit_cast(__bf16, h);
}
#if __has_builtin(__builtin_amdgcn_tanhf)
__device__ __forceinline__ float tanh_(float x) { return __builtin_amdgcn_tanhf(x); }  // v_tanh_f32
#else
__device__ __forceinline__ float tanh_(float x) { return tanhf(x); }
#endif
__device__ __forceinline__ float sig_(float x) {
  return __builtin_amdgcn_rcpf(1.f + __expf(-x));  // v_rcp_f32 + v_exp_f32
}

#define WMMA_BF16(acc, a, b) \
  __builtin_amdgcn_wmma_f32_16x16x32_bf16(false, (a), false, (b), (short)0, (acc), false, false)

// ---------- fragment helpers ----------
// A-matrix 16x32 bf16 (ISA 7.12.2): lane&15=row; lane<16 -> K in {0..7, 16..23}; lane>=16 -> +8.
// Packed A storage: frag(kt) -> 32 lanes x 16 contiguous bf16.
__device__ __forceinline__ v16bf load_a_pack(const unsigned short* pk, int kt, int lane) {
  return *(const v16bf*)(pk + (((kt << 5) + lane) << 4));
}
// inverse map: element (row b, K index d) -> packed offset
__device__ __forceinline__ int apack_idx(int b, int d) {
  int kt   = d >> 5, r = d & 31;
  int lane = b + ((r & 8) ? 16 : 0);
  int e    = (r & 7) + ((r & 16) ? 8 : 0);
  return ((kt << 5) + lane) * 16 + e;
}
// A fragment from f32 row-major (used by one-time GEMMs only)
__device__ __forceinline__ v16bf load_a_f32(const float* base, int ldk, int kt, int lane) {
  int row = lane & 15;
  int kb  = kt * 32 + ((lane & 16) ? 8 : 0);
  const float* p = base + row * ldk + kb;
  v16bf a;
#pragma unroll
  for (int e = 0; e < 8; e++) {
    a[e]     = bfbits(f2bf(p[e]));
    a[e + 8] = bfbits(f2bf(p[16 + e]));
  }
  return a;
}
__device__ __forceinline__ v16bf load_a_bf16(const unsigned short* base, int ldk, int kt, int lane) {
  int row = lane & 15;
  int kb  = kt * 32 + ((lane & 16) ? 8 : 0);
  const unsigned short* p = base + row * ldk + kb;
  v16bf a;
#pragma unroll
  for (int e = 0; e < 8; e++) {
    a[e]     = bfbits(p[e]);
    a[e + 8] = bfbits(p[16 + e]);
  }
  return a;
}
// B fragments pre-packed: frag -> 32 lanes x 16 contiguous bf16
__device__ __forceinline__ v16bf load_b_pack(const unsigned short* pack, int frag, int lane) {
  return *(const v16bf*)(pack + (((frag << 5) + lane) << 4));
}

// ---------- one-time: pack f32 KxN weight into WMMA B-fragment order ----------
__global__ void pack_b_kernel(const float* __restrict__ src, int Kt, int Nt,
                              unsigned short* __restrict__ dst) {
  int idx   = blockIdx.x * blockDim.x + threadIdx.x;
  int total = Kt * Nt * 32;
  if (idx >= total) return;
  int lane = idx & 31;
  int frag = idx >> 5;
  int nt   = frag % Nt;
  int kt   = frag / Nt;
  int Ntot = Nt * 16;
  int col  = nt * 16 + (lane & 15);
  int kb   = kt * 32 + ((lane & 16) ? 8 : 0);
  unsigned short* d = dst + idx * 16;
#pragma unroll
  for (int e = 0; e < 16; e++) d[e] = f2bf(src[(kb + e + (e & 8)) * Ntot + col]);
}

__global__ void cvt_bf16_kernel(const float* __restrict__ src, unsigned short* __restrict__ dst, int n) {
  for (int i = blockIdx.x * blockDim.x + threadIdx.x; i < n; i += gridDim.x * blockDim.x)
    dst[i] = f2bf(src[i]);
}

// ---------- enc_proj = X @ We  (16384x256 @ 256x256) -> bf16 row-major ----------
__global__ void enc_gemm_kernel(const float* __restrict__ X, const unsigned short* __restrict__ WeP,
                                unsigned short* __restrict__ encp) {
  int lane = threadIdx.x & 31;
  int wid  = threadIdx.x >> 5;
  int tile = blockIdx.x * 4 + wid;  // 16384 tiles
  int nt   = tile & 15;
  int mt   = tile >> 4;
  const float* arow = X + mt * 16 * EE;
  v8f acc = {};
#pragma unroll
  for (int kt = 0; kt < 8; kt++) {
    v16bf a = load_a_f32(arow, EE, kt, lane);
    v16bf b = load_b_pack(WeP, kt * 16 + nt, lane);
    acc = WMMA_BF16(acc, a, b);
  }
  int col = nt * 16 + (lane & 15);
  int rb  = mt * 16 + ((lane & 16) ? 8 : 0);
#pragma unroll
  for (int v = 0; v < 8; v++) encp[(rb + v) * EE + col] = f2bf(acc[v]);
}

// ---------- final: out = [h_T, ctx_T] @ fc_out_W + b  (128x512 @ 512x8192) ----------
__global__ void out_gemm_kernel(const unsigned short* __restrict__ A,
                                const unsigned short* __restrict__ Bp,
                                const float* __restrict__ bias, float* __restrict__ out) {
  int lane = threadIdx.x & 31;
  int wid  = threadIdx.x >> 5;
  int tile = blockIdx.x * 4 + wid;  // 4096 tiles
  int nt   = tile & 511;
  int mt   = tile >> 9;
  const unsigned short* arow = A + mt * 16 * 512;
  v8f acc = {};
#pragma unroll
  for (int kt = 0; kt < 16; kt++) {
    v16bf a = load_a_bf16(arow, 512, kt, lane);
    v16bf b = load_b_pack(Bp, kt * 512 + nt, lane);
    acc = WMMA_BF16(acc, a, b);
  }
  int   col = nt * 16 + (lane & 15);
  int   rb  = mt * 16 + ((lane & 16) ? 8 : 0);
  float bb  = bias[col];
#pragma unroll
  for (int v = 0; v < 8; v++) out[(rb + v) * 8192 + col] = acc[v] + bb;
}

// ---------- A fragment from y_history[b, t, :] (f32, B x S x OUT) ----------
__device__ __forceinline__ v16bf load_a_yh(const float* __restrict__ Y, int bg0, int t, int kt, int lane) {
  int b  = bg0 + (lane & 15);
  int kb = kt * 32 + ((lane & 16) ? 8 : 0);
  const float* p = Y + (b * SS + t) * OUTN + kb;
  v16bf a;
#pragma unroll
  for (int e = 0; e < 8; e++) {
    a[e]     = bfbits(f2bf(p[e]));
    a[e + 8] = bfbits(f2bf(p[16 + e]));
  }
  return a;
}

// ---------- persistent scan: one block = 16 batch rows through all 128 steps ----------
// 16 waves (512 threads). LDS (bytes):
//   f32: h 16K | c 16K | hc 16K | ctx 16K | alpha 8K            = 73,728
//   bf16 packed A: ha 8K | ca 8K | ctxa 8K | yta 2K             = 26,624
//   total 100,352
__global__ void scan_kernel(const float* __restrict__ h0, const float* __restrict__ c0,
                            const unsigned short* __restrict__ encp,
                            const unsigned short* __restrict__ xbf,
                            const float* __restrict__ Yh,
                            const float* __restrict__ w2, const float* __restrict__ b2p,
                            const float* __restrict__ b1,
                            const unsigned short* __restrict__ WhP,
                            const unsigned short* __restrict__ WcP,
                            const unsigned short* __restrict__ fcWP, const float* __restrict__ fcb,
                            const unsigned short* __restrict__ WihP,
                            const unsigned short* __restrict__ WhhP,
                            const float* __restrict__ bih, const float* __restrict__ bhh,
                            unsigned short* __restrict__ hctx) {
  extern __shared__ float smem[];
  float* sh_h   = smem;            // 16*256
  float* sh_c   = sh_h + 4096;     // 16*256
  float* sh_hc  = sh_c + 4096;     // 16*256
  float* sh_ctx = sh_hc + 4096;    // 16*256
  float* sh_al  = sh_ctx + 4096;   // 16*128
  unsigned short* sh_ha   = (unsigned short*)(sh_al + 2048);  // packed A: 8 kt frags
  unsigned short* sh_ca   = sh_ha + 4096;
  unsigned short* sh_ctxa = sh_ca + 4096;
  unsigned short* sh_yta  = sh_ctxa + 4096;                   // 2 kt frags (K=64)

  const int   tid  = threadIdx.x;
  const int   lane = tid & 31;
  const int   wid  = tid >> 5;
  const int   bg0  = blockIdx.x * 16;
  const float b2   = b2p[0];

  // ---- loop-invariant register state ----
  const int   dcol = wid * 16 + (lane & 15);   // this wave's hidden-dim column (0..255)
  const int   rb   = (lane & 16) ? 8 : 0;
  float       w2v[8];
#pragma unroll
  for (int j = 0; j < 8; j++) w2v[j] = w2[j * 32 + lane];
  const float hcbias = b1[dcol];
  const float ytbias = (wid < 4) ? fcb[wid * 16 + (lane & 15)] : 0.f;
  const float bI = bih[dcol] + bhh[dcol];
  const float bF = bih[256 + dcol] + bhh[256 + dcol];
  const float bG = bih[512 + dcol] + bhh[512 + dcol];
  const float bO = bih[768 + dcol] + bhh[768 + dcol];

  // ---- init h, c (f32 + packed bf16) ----
  for (int i = tid; i < 4096; i += 512) {
    int   b = i >> 8, d = i & 255;
    float hv = h0[(bg0 + b) * DD + d];
    float cv = c0[(bg0 + b) * DD + d];
    sh_h[i] = hv;
    sh_c[i] = cv;
    int pi  = apack_idx(b, d);
    sh_ha[pi] = f2bf(hv);
    sh_ca[pi] = f2bf(cv);
  }
  __syncthreads();

  for (int t = 0; t < SS; t++) {
    // ---- 1. hc = h@Wh + c@Wc + b1 : wave w owns N-tile w ----
    {
      v8f acc = {};
#pragma unroll
      for (int kt = 0; kt < 8; kt++) {
        v16bf a = load_a_pack(sh_ha, kt, lane);
        v16bf b = load_b_pack(WhP, kt * 16 + wid, lane);
        acc = WMMA_BF16(acc, a, b);
      }
#pragma unroll
      for (int kt = 0; kt < 8; kt++) {
        v16bf a = load_a_pack(sh_ca, kt, lane);
        v16bf b = load_b_pack(WcP, kt * 16 + wid, lane);
        acc = WMMA_BF16(acc, a, b);
      }
#pragma unroll
      for (int v = 0; v < 8; v++) sh_hc[(rb + v) * EE + dcol] = acc[v] + hcbias;
    }
    __syncthreads();

    // ---- 2. scores[b,s] : wave w owns batch row b=w ----
    {
      int   b  = wid;
      int   bg = bg0 + b;
      float hcv[8];
#pragma unroll
      for (int j = 0; j < 8; j++) hcv[j] = sh_hc[b * EE + j * 32 + lane];
      for (int s = 0; s < SS; s++) {
        const unsigned short* ep = encp + (bg * SS + s) * EE;
        if (s + 1 < SS) __builtin_prefetch(ep + EE, 0, 0);  // global_prefetch_b8
        float part = 0.f;
#pragma unroll
        for (int j = 0; j < 8; j++)
          part += w2v[j] * tanh_(bf2f(ep[j * 32 + lane]) + hcv[j]);
#pragma unroll
        for (int off = 16; off > 0; off >>= 1) part += __shfl_xor(part, off, 32);
        if (lane == 0) sh_al[b * SS + s] = part + b2;
      }
    }
    __syncthreads();

    // ---- 3. softmax over s : wave w owns row b=w ----
    {
      int   b = wid;
      float v0[4], m = -3.0e38f;
#pragma unroll
      for (int k = 0; k < 4; k++) { v0[k] = sh_al[b * SS + lane + k * 32]; m = fmaxf(m, v0[k]); }
#pragma unroll
      for (int off = 16; off > 0; off >>= 1) m = fmaxf(m, __shfl_xor(m, off, 32));
      float sum = 0.f;
#pragma unroll
      for (int k = 0; k < 4; k++) { v0[k] = __expf(v0[k] - m); sum += v0[k]; }
#pragma unroll
      for (int off = 16; off > 0; off >>= 1) sum += __shfl_xor(sum, off, 32);
      float inv = __builtin_amdgcn_rcpf(sum);
#pragma unroll
      for (int k = 0; k < 4; k++) sh_al[b * SS + lane + k * 32] = v0[k] * inv;
    }
    __syncthreads();

    // ---- 4. context[b,e] = sum_s alpha * X[b,s,e] (bf16 X) ----
    {
      int e  = tid & 255;
      int bh = tid >> 8;
      for (int pass = 0; pass < 8; pass++) {
        int b = pass * 2 + bh;
        const unsigned short* xp = xbf + ((bg0 + b) * SS) * EE + e;
        float acc = 0.f;
        for (int s = 0; s < SS; s++) acc += sh_al[b * SS + s] * bf2f(xp[s * EE]);
        sh_ctx[b * EE + e]          = acc;
        sh_ctxa[apack_idx(b, e)]    = f2bf(acc);
      }
    }
    __syncthreads();

    // ---- 5. y_tilde = [ctx, y_t]@fc_W + fc_b : waves 0..3, store packed-A ----
    if (wid < 4) {
      v8f acc = {};
#pragma unroll
      for (int kt = 0; kt < 8; kt++) {
        v16bf a = load_a_pack(sh_ctxa, kt, lane);
        v16bf b = load_b_pack(fcWP, kt * 4 + wid, lane);
        acc = WMMA_BF16(acc, a, b);
      }
#pragma unroll
      for (int kt = 0; kt < 2; kt++) {
        v16bf a = load_a_yh(Yh, bg0, t, kt, lane);
        v16bf b = load_b_pack(fcWP, (8 + kt) * 4 + wid, lane);
        acc = WMMA_BF16(acc, a, b);
      }
      int col = wid * 16 + (lane & 15);
#pragma unroll
      for (int v = 0; v < 8; v++) sh_yta[apack_idx(rb + v, col)] = f2bf(acc[v] + ytbias);
    }
    __syncthreads();

    // ---- 6. gates GEMM fused with LSTM pointwise ----
    // wave w's N-tiles {w, w+16, w+32, w+48} are exactly the i/f/g/o slices for
    // hidden dims [16w, 16w+16): keep 4 accumulators, apply activations in regs.
    {
      v16bf a_yt[2], a_h[8];
#pragma unroll
      for (int kt = 0; kt < 2; kt++) a_yt[kt] = load_a_pack(sh_yta, kt, lane);
#pragma unroll
      for (int kt = 0; kt < 8; kt++) a_h[kt] = load_a_pack(sh_ha, kt, lane);
      __syncthreads();  // all reads of sh_ha/sh_yta done before overwrites below

      v8f acc[4] = {{}, {}, {}, {}};
#pragma unroll
      for (int g = 0; g < 4; g++) {
        int nt = wid + g * 16;
#pragma unroll
        for (int kt = 0; kt < 2; kt++) {
          v16bf b = load_b_pack(WihP, kt * 64 + nt, lane);
          acc[g]  = WMMA_BF16(acc[g], a_yt[kt], b);
        }
#pragma unroll
        for (int kt = 0; kt < 8; kt++) {
          v16bf b = load_b_pack(WhhP, kt * 64 + nt, lane);
          acc[g]  = WMMA_BF16(acc[g], a_h[kt], b);
        }
      }
#pragma unroll
      for (int v = 0; v < 8; v++) {
        int   b  = rb + v;
        float ii = sig_(acc[0][v] + bI);
        float ff = sig_(acc[1][v] + bF);
        float gg = tanh_(acc[2][v] + bG);
        float oo = sig_(acc[3][v] + bO);
        float cn = ff * sh_c[b * DD + dcol] + ii * gg;
        float hn = oo * tanh_(cn);
        sh_c[b * DD + dcol] = cn;
        sh_h[b * DD + dcol] = hn;
        int pi = apack_idx(b, dcol);
        sh_ca[pi] = f2bf(cn);
        sh_ha[pi] = f2bf(hn);
      }
    }
    __syncthreads();
  }

  // ---- write [h_T, ctx_T] as bf16 (B x 512) for final GEMM ----
  for (int i = tid; i < 16 * 512; i += 512) {
    int   b = i >> 9, j = i & 511;
    float v = (j < 256) ? sh_h[b * DD + j] : sh_ctx[b * EE + (j - 256)];
    hctx[(bg0 + b) * 512 + j] = f2bf(v);
  }
}

// ---------- workspace layout (unsigned short element offsets) ----------
#define WH_P    0u          // 8x16  frags  (65536)
#define WC_P    65536u      // 8x16         (65536)
#define WE_P    131072u     // 8x16         (65536)
#define WHH_P   196608u     // 8x64         (262144)
#define WIH_P   458752u     // 2x64         (65536)
#define FCW_P   524288u     // 10x4         (20480)
#define FCOUT_P 544768u     // 16x512       (4194304)
#define ENC_P   4739072u    // enc_proj bf16 row-major (4194304)
#define XBF_P   8933376u    // X bf16 row-major        (4194304)
#define HCTX_P  13127680u   // [h,ctx] bf16 (65536)

#define SCAN_SMEM 100352

extern "C" void kernel_launch(void* const* d_in, const int* in_sizes, int n_in,
                              void* d_out, int out_size, void* d_ws, size_t ws_size,
                              hipStream_t stream) {
  const float* X    = (const float*)d_in[0];
  const float* Yh   = (const float*)d_in[1];
  const float* h0   = (const float*)d_in[2];
  const float* c0   = (const float*)d_in[3];
  const float* W1   = (const float*)d_in[4];   // (768, 256): Wh | Wc | We
  const float* b1   = (const float*)d_in[5];
  const float* w2   = (const float*)d_in[6];
  const float* b2   = (const float*)d_in[7];
  const float* Wih  = (const float*)d_in[8];   // (64, 1024)
  const float* Whh  = (const float*)d_in[9];   // (256, 1024)
  const float* bih  = (const float*)d_in[10];
  const float* bhh  = (const float*)d_in[11];
  const float* fcW  = (const float*)d_in[12];  // (320, 64)
  const float* fcb  = (const float*)d_in[13];
  const float* foW  = (const float*)d_in[14];  // (512, 8192)
  const float* fob  = (const float*)d_in[15];
  unsigned short* ws = (unsigned short*)d_ws;
  float* out = (float*)d_out;

  // one-time weight packing to WMMA B-fragment order (bf16)
  pack_b_kernel<<<16, 256, 0, stream>>>(W1,             8, 16,  ws + WH_P);
  pack_b_kernel<<<16, 256, 0, stream>>>(W1 + 256 * 256, 8, 16,  ws + WC_P);
  pack_b_kernel<<<16, 256, 0, stream>>>(W1 + 512 * 256, 8, 16,  ws + WE_P);
  pack_b_kernel<<<64, 256, 0, stream>>>(Whh,            8, 64,  ws + WHH_P);
  pack_b_kernel<<<16, 256, 0, stream>>>(Wih,            2, 64,  ws + WIH_P);
  pack_b_kernel<<<5,  256, 0, stream>>>(fcW,           10, 4,   ws + FCW_P);
  pack_b_kernel<<<1024, 256, 0, stream>>>(foW,         16, 512, ws + FCOUT_P);

  // input_encoded -> bf16 (context pass streams this every step)
  cvt_bf16_kernel<<<2048, 256, 0, stream>>>(X, ws + XBF_P, BB * SS * EE);

  // enc_proj = X @ We  (WMMA), stored bf16
  enc_gemm_kernel<<<4096, 128, 0, stream>>>(X, ws + WE_P, ws + ENC_P);

  // persistent 128-step scan: 8 workgroups x 512 threads, ~100KB LDS each
  hipFuncSetAttribute((const void*)scan_kernel,
                      hipFuncAttributeMaxDynamicSharedMemorySize, SCAN_SMEM);
  scan_kernel<<<BB / 16, 512, SCAN_SMEM, stream>>>(
      h0, c0, ws + ENC_P, ws + XBF_P, Yh, w2, b2, b1,
      ws + WH_P, ws + WC_P, ws + FCW_P, fcb, ws + WIH_P, ws + WHH_P,
      bih, bhh, ws + HCTX_P);

  // out = [h_T, ctx_T] @ fc_out_W + fc_out_b  (WMMA)
  out_gemm_kernel<<<1024, 128, 0, stream>>>(ws + HCTX_P, ws + FCOUT_P, fob, out);
}
